// TransformerBlock_59966333386934
// MI455X (gfx1250) — compile-verified, matrix operand order checked by reading, not run on hardware
//
#include <hip/hip_runtime.h>
#include <hip/hip_bf16.h>
#include <math.h>

typedef __attribute__((ext_vector_type(16))) _Float16 v16h;
typedef __attribute__((ext_vector_type(8)))  float    v8f;

#define S_LEN   2048
#define E_DIM   1024
#define H_NUM   16
#define D_DIM   32
#define SIG_DIM 256
#define LAMBDA_INIT_F 0.3555090675909693f

__device__ __forceinline__ v8f wmma16x16x32(v16h a, v16h b, v8f c) {
  // v_wmma_f32_16x16x32_f16: D = A(16x32,f16) x B(32x16,f16) + C(f32)
  return __builtin_amdgcn_wmma_f32_16x16x32_f16(false, a, false, b, (short)0, c, false, false);
}

// ---------------------------------------------------------------- conversions
__global__ void cvt_f32_f16(const float* __restrict__ src, _Float16* __restrict__ dst,
                            long long n) {
  long long i = (long long)blockIdx.x * blockDim.x + threadIdx.x;
  long long stride = (long long)gridDim.x * blockDim.x;
  for (; i < n; i += stride) dst[i] = (_Float16)src[i];
}

// --------------------------------------------------- sig = emb @ sigma_w.T ; lambda
__global__ void small_setup(const float* __restrict__ emb, const float* __restrict__ sigw,
                            const float* __restrict__ lq1, const float* __restrict__ lk1,
                            const float* __restrict__ lq2, const float* __restrict__ lk2,
                            float* __restrict__ sig, float* __restrict__ lam) {
  int t = threadIdx.x;
  if (t < D_DIM) {
    float s = 0.f;
    for (int j = 0; j < SIG_DIM; ++j) s += emb[j] * sigw[t * SIG_DIM + j];
    sig[t] = s;
  } else if (t == D_DIM) {
    float s1 = 0.f, s2 = 0.f;
    for (int i = 0; i < D_DIM; ++i) { s1 += lq1[i] * lk1[i]; s2 += lq2[i] * lk2[i]; }
    lam[0] = __expf(s1) - __expf(s2) + LAMBDA_INIT_F;
  }
}

// ------------------------------------------------------------------- GEMM
// C[M,N] = A[M,K] * B[N,K]^T (+bias). MODE 0: fp32 out. MODE 1: SiLU -> f16 out.
// Double-buffered LDS, global_prefetch of K+2 tile, branch-free templated epilogue.
#define BM 128
#define BN 128
#define BK 32
#define ASTR 40   // LDS stride in halfs (padded)

template <int MODE, bool HAS_BIAS>
__global__ __launch_bounds__(256)
void gemm_wmma(const _Float16* __restrict__ A, const _Float16* __restrict__ B,
               const float* __restrict__ bias, float* __restrict__ Cf,
               _Float16* __restrict__ Ch, int M, int N, int K) {
  __shared__ _Float16 As[2][BM * ASTR];
  __shared__ _Float16 Bs[2][BN * ASTR];
  const int tid   = threadIdx.x;
  const int lane  = tid & 31;
  const int wave  = tid >> 5;      // 8 wave32 waves
  const int wm    = wave >> 2;     // 0..1  (64 rows each)
  const int wn    = wave & 3;      // 0..3  (32 cols each)
  const int l15   = lane & 15;
  const int hioff = (lane >> 4) * 8;
  const int bm = blockIdx.y * BM;
  const int bn = blockIdx.x * BN;

  const v8f vzero = {0.f,0.f,0.f,0.f,0.f,0.f,0.f,0.f};
  v8f acc[4][2];
  #pragma unroll
  for (int mt = 0; mt < 4; ++mt)
    #pragma unroll
    for (int nt = 0; nt < 2; ++nt) acc[mt][nt] = vzero;

  const int ldRow = tid >> 1;           // 0..127
  const int ldCol = (tid & 1) * 16;     // 0 or 16 halfs
  const _Float16* gA = A + (size_t)(bm + ldRow) * K + ldCol;
  const _Float16* gB = B + (size_t)(bn + ldRow) * K + ldCol;

  // preload tile 0 into buffer 0
  {
    const uint4* sa = (const uint4*)gA;
    const uint4* sb = (const uint4*)gB;
    uint4* da = (uint4*)(&As[0][ldRow * ASTR + ldCol]);
    uint4* db = (uint4*)(&Bs[0][ldRow * ASTR + ldCol]);
    da[0] = sa[0]; da[1] = sa[1];
    db[0] = sb[0]; db[1] = sb[1];
  }
  __syncthreads();

  const int ktiles = K / BK;
  int cur = 0;
  for (int kt = 0; kt < ktiles; ++kt) {
    // pull the K+2 tile toward L2 (global_prefetch_b8)
    if (kt + 2 < ktiles) {
      __builtin_prefetch(gA + (size_t)(kt + 2) * BK, 0, 1);
      __builtin_prefetch(gB + (size_t)(kt + 2) * BK, 0, 1);
    }
    // stage K+1 tile into the alternate buffer while we compute from `cur`
    if (kt + 1 < ktiles) {
      const uint4* sa = (const uint4*)(gA + (size_t)(kt + 1) * BK);
      const uint4* sb = (const uint4*)(gB + (size_t)(kt + 1) * BK);
      const uint4 ra0 = sa[0], ra1 = sa[1], rb0 = sb[0], rb1 = sb[1];
      uint4* da = (uint4*)(&As[cur ^ 1][ldRow * ASTR + ldCol]);
      uint4* db = (uint4*)(&Bs[cur ^ 1][ldRow * ASTR + ldCol]);
      da[0] = ra0; da[1] = ra1;
      db[0] = rb0; db[1] = rb1;
    }

    v16h af[4], bf2[2];
    #pragma unroll
    for (int mt = 0; mt < 4; ++mt) {
      const _Float16* p = &As[cur][(wm * 64 + mt * 16 + l15) * ASTR + hioff];
      #pragma unroll
      for (int i = 0; i < 8; ++i) { af[mt][i] = p[i]; af[mt][8 + i] = p[16 + i]; }
    }
    #pragma unroll
    for (int nt = 0; nt < 2; ++nt) {
      const _Float16* p = &Bs[cur][(wn * 32 + nt * 16 + l15) * ASTR + hioff];
      #pragma unroll
      for (int i = 0; i < 8; ++i) { bf2[nt][i] = p[i]; bf2[nt][8 + i] = p[16 + i]; }
    }
    #pragma unroll
    for (int mt = 0; mt < 4; ++mt)
      #pragma unroll
      for (int nt = 0; nt < 2; ++nt)
        acc[mt][nt] = wmma16x16x32(af[mt], bf2[nt], acc[mt][nt]);

    __syncthreads();   // next-tile LDS writes complete; safe to flip buffers
    cur ^= 1;
  }

  // ---------------- branch-free epilogue (bias hoisted, strided stores)
  float bv[2] = {0.f, 0.f};
  if (HAS_BIAS) {
    bv[0] = bias[bn + wn * 32 + l15];
    bv[1] = bias[bn + wn * 32 + 16 + l15];
  }
  #pragma unroll
  for (int mt = 0; mt < 4; ++mt) {
    const int gm0 = bm + wm * 64 + mt * 16 + ((lane >> 4) << 3);
    #pragma unroll
    for (int nt = 0; nt < 2; ++nt) {
      const int gn = bn + wn * 32 + nt * 16 + l15;
      if (MODE == 1) {
        _Float16* cp = Ch + (size_t)gm0 * N + gn;
        #pragma unroll
        for (int r = 0; r < 8; ++r) {
          float v = acc[mt][nt][r];
          if (HAS_BIAS) v += bv[nt];
          cp[(size_t)r * N] = (_Float16)(v / (1.f + __expf(-v)));   // SiLU
        }
      } else {
        float* cp = Cf + (size_t)gm0 * N + gn;
        #pragma unroll
        for (int r = 0; r < 8; ++r) {
          float v = acc[mt][nt][r];
          if (HAS_BIAS) v += bv[nt];
          cp[(size_t)r * N] = v;
        }
      }
    }
  }
}

// ------------------------------------------- RoPE + sigma add + head split -> f16
// Q/K written [h][s][d] (row = K-contig for A/B frags).
// V written TRANSPOSED [h][vdim][s] so attention's B-frag of V is K(seq)-contiguous.
__global__ __launch_bounds__(256)
void rope_split(const float* __restrict__ qkv, const float* __restrict__ sig,
                _Float16* __restrict__ q1, _Float16* __restrict__ q2,
                _Float16* __restrict__ k1, _Float16* __restrict__ k2,
                _Float16* __restrict__ vt) {
  const int s = blockIdx.x;
  const int t = threadIdx.x;
  __shared__ float sigs[D_DIM];
  if (t < D_DIM) sigs[t] = sig[t];
  __syncthreads();
  const float* row = qkv + (size_t)s * (3 * E_DIM);
  for (int idx = t; idx < 3 * E_DIM; idx += 256) {
    const int which = idx >> 10;        // 0=q 1=k 2=v
    const int rem   = idx & 1023;
    const int rh    = rem >> 5;         // rope head 0..31
    const int d     = rem & 31;
    const int h     = rh >> 1;
    const int sub   = rh & 1;
    const float val = row[idx];
    float o = val;
    if (which < 2 && d < 16) {          // rotate first half of head dim, pairwise
      const int p = d >> 1;
      const float inv = __powf(10000.f, -(float)(2 * p) * (1.f / 16.f));
      const float ang = (float)s * inv;
      float sn, c; __sincosf(ang, &sn, &c);
      const float other = row[(idx & ~31) + (d ^ 1)];
      o = (d & 1) ? (val * c + other * sn) : (val * c - other * sn);
    }
    if (which < 2 && sub == 1) o += sigs[d];
    if (which == 0) {
      _Float16* dst = sub ? q2 : q1;
      dst[(size_t)h * S_LEN * D_DIM + (size_t)s * D_DIM + d] = (_Float16)o;
    } else if (which == 1) {
      _Float16* dst = sub ? k2 : k1;
      dst[(size_t)h * S_LEN * D_DIM + (size_t)s * D_DIM + d] = (_Float16)o;
    } else {
      // transposed: [h][vdim = sub*32+d][s]
      vt[(size_t)h * (2 * D_DIM) * S_LEN + (size_t)(sub * D_DIM + d) * S_LEN + s] = (_Float16)o;
    }
  }
}

// ----------------------------------------------------- flash attention (WMMA)
#define KSTR 40
#define VSTR 72   // Vs rows are vdim (64), cols are seq/K (64) + pad
#define PSTR 72

__global__ __launch_bounds__(128)
void attn_flash(const _Float16* __restrict__ q1, const _Float16* __restrict__ q2,
                const _Float16* __restrict__ k1, const _Float16* __restrict__ k2,
                const _Float16* __restrict__ vt,
                float* __restrict__ o1, float* __restrict__ o2, float scale) {
  __shared__ _Float16 Ks[64 * KSTR];
  __shared__ _Float16 Vs[64 * VSTR];    // transposed V tile: [vdim][seq]
  __shared__ _Float16 Ps[4 * 16 * PSTR];

  const int tid   = threadIdx.x;
  const int lane  = tid & 31;
  const int wave  = tid >> 5;          // 4 waves, 16 q-rows each
  const int l15   = lane & 15;
  const int hi    = lane >> 4;
  const int hioff = hi * 8;

  const int qi = blockIdx.x;           // q tile (64 rows)
  const int h  = blockIdx.y;
  const int ai = blockIdx.z;           // which attention (0/1)

  const _Float16* Qp = (ai ? q2 : q1) + (size_t)h * S_LEN * D_DIM;
  const _Float16* Kp = (ai ? k2 : k1) + (size_t)h * S_LEN * D_DIM;
  const _Float16* Vp = vt + (size_t)h * (2 * D_DIM) * S_LEN;   // [vdim][s]
  float* Op = ai ? o2 : o1;

  const int qbase = qi * 64;
  const int qrow0 = qbase + wave * 16;

  v16h aq;                              // Q A-frag (rows contiguous in global, K=32)
  {
    const _Float16* p = Qp + (size_t)(qrow0 + l15) * D_DIM + hioff;
    #pragma unroll
    for (int i = 0; i < 8; ++i) { aq[i] = p[i]; aq[8 + i] = p[16 + i]; }
  }

  const v8f vzero = {0.f,0.f,0.f,0.f,0.f,0.f,0.f,0.f};
  v8f o[4];
  #pragma unroll
  for (int j = 0; j < 4; ++j) o[j] = vzero;
  float rmax[8], rsum[8];
  #pragma unroll
  for (int r = 0; r < 8; ++r) { rmax[r] = -1e30f; rsum[r] = 0.f; }

  const int ldRow = tid >> 1;           // 0..63
  const int kc16  = (tid & 1) * 16;
  const int vc32  = (tid & 1) * 32;

  for (int kt = 0; kt <= qi; ++kt) {
    const int kbase = kt * 64;
    {   // stage K (64 seq x 32 d) and V^T (64 vdim x 64 seq) tiles, all vectorized
      const uint4* s = (const uint4*)(Kp + (size_t)(kbase + ldRow) * D_DIM + kc16);
      uint4* d = (uint4*)(Ks + ldRow * KSTR + kc16);
      d[0] = s[0]; d[1] = s[1];
      const uint4* s2 = (const uint4*)(Vp + (size_t)ldRow * S_LEN + kbase + vc32);
      uint4* d2 = (uint4*)(Vs + ldRow * VSTR + vc32);
      d2[0] = s2[0]; d2[1] = s2[1]; d2[2] = s2[2]; d2[3] = s2[3];
    }
    __syncthreads();

    // scores = Q(16x32) x K^T(32x64): 4 WMMA
    v8f sc[4];
    #pragma unroll
    for (int j = 0; j < 4; ++j) {
      v16h bk;
      const _Float16* p = Ks + (j * 16 + l15) * KSTR + hioff;
      #pragma unroll
      for (int i = 0; i < 8; ++i) { bk[i] = p[i]; bk[8 + i] = p[16 + i]; }
      sc[j] = wmma16x16x32(aq, bk, vzero);
      #pragma unroll
      for (int r = 0; r < 8; ++r) sc[j][r] *= scale;
    }
    if (kt == qi) {                      // causal mask on diagonal tile
      #pragma unroll
      for (int j = 0; j < 4; ++j) {
        const int col = kbase + j * 16 + l15;
        #pragma unroll
        for (int r = 0; r < 8; ++r) {
          const int rowg = qrow0 + r + (hi << 3);
          if (col > rowg) sc[j][r] = -1e30f;
        }
      }
    }

    // online softmax (row stats live in VGPR index r, reduced over 16-lane groups)
    float nmax[8], alpha[8], psum[8];
    #pragma unroll
    for (int r = 0; r < 8; ++r) {
      float m = sc[0][r];
      #pragma unroll
      for (int j = 1; j < 4; ++j) m = fmaxf(m, sc[j][r]);
      #pragma unroll
      for (int off2 = 8; off2 > 0; off2 >>= 1) m = fmaxf(m, __shfl_xor(m, off2, 16));
      nmax[r] = fmaxf(rmax[r], m);
      alpha[r] = __expf(rmax[r] - nmax[r]);
      rmax[r] = nmax[r];
      psum[r] = 0.f;
    }
    #pragma unroll
    for (int j = 0; j < 4; ++j)
      #pragma unroll
      for (int r = 0; r < 8; ++r) {
        const float p = __expf(sc[j][r] - nmax[r]);
        sc[j][r] = p;
        psum[r] += p;
      }
    #pragma unroll
    for (int r = 0; r < 8; ++r) {
      float ps = psum[r];
      #pragma unroll
      for (int off2 = 8; off2 > 0; off2 >>= 1) ps += __shfl_xor(ps, off2, 16);
      rsum[r] = rsum[r] * alpha[r] + ps;
    }
    #pragma unroll
    for (int j = 0; j < 4; ++j)
      #pragma unroll
      for (int r = 0; r < 8; ++r) o[j][r] *= alpha[r];

    // P (C/D layout) -> LDS -> A-frag layout
    _Float16* Pw = Ps + wave * 16 * PSTR;
    #pragma unroll
    for (int j = 0; j < 4; ++j)
      #pragma unroll
      for (int r = 0; r < 8; ++r)
        Pw[(r + (hi << 3)) * PSTR + j * 16 + l15] = (_Float16)sc[j][r];
    __syncthreads();

    v16h ap0, ap1;
    {
      const _Float16* p = Pw + l15 * PSTR;
      #pragma unroll
      for (int i = 0; i < 8; ++i) {
        ap0[i]     = p[hioff + i];       ap0[8 + i] = p[16 + hioff + i];
        ap1[i]     = p[32 + hioff + i];  ap1[8 + i] = p[48 + hioff + i];
      }
    }
    // O += P(16x64) x V(64x64): 8 WMMA.
    // V^T layout makes each B-frag chunk 8 contiguous halfs -> ds_load_b128.
    #pragma unroll
    for (int j = 0; j < 4; ++j) {
      const _Float16* pv = Vs + (j * 16 + l15) * VSTR;
      v16h bv;
      #pragma unroll
      for (int i = 0; i < 8; ++i) { bv[i] = pv[hioff + i]; bv[8 + i] = pv[16 + hioff + i]; }
      o[j] = wmma16x16x32(ap0, bv, o[j]);
      #pragma unroll
      for (int i = 0; i < 8; ++i) { bv[i] = pv[32 + hioff + i]; bv[8 + i] = pv[48 + hioff + i]; }
      o[j] = wmma16x16x32(ap1, bv, o[j]);
    }
    __syncthreads();
  }

  // finalize: strided stores from a hoisted base pointer
  #pragma unroll
  for (int r = 0; r < 8; ++r) {
    const float invs = 1.f / rsum[r];
    float* rp = Op + (size_t)(qrow0 + r + (hi << 3)) * E_DIM + h * 64 + l15;
    #pragma unroll
    for (int j = 0; j < 4; ++j)
      rp[j * 16] = o[j][r] * invs;
  }
}

// -------------------------------- a = rmsnorm(attn1 - lam*attn2)*(1-LAMBDA) -> f16
__global__ __launch_bounds__(64)
void diff_rmsnorm(const float* __restrict__ a1, const float* __restrict__ a2,
                  const float* __restrict__ lamp, const float* __restrict__ w,
                  const float* __restrict__ b, _Float16* __restrict__ outh) {
  const int idx = blockIdx.x;
  const int s = idx >> 4;
  const int h = idx & 15;
  const int t = threadIdx.x;                    // 0..63 over the 2D-dim group
  __shared__ float sh[2];
  const float lam = lamp[0];
  const size_t off = (size_t)s * E_DIM + h * 64 + t;
  const float d = a1[off] - lam * a2[off];
  float sq = d * d;
  #pragma unroll
  for (int o = 16; o > 0; o >>= 1) sq += __shfl_xor(sq, o, 32);
  if ((t & 31) == 0) sh[t >> 5] = sq;
  __syncthreads();
  const float ms = (sh[0] + sh[1]) * (1.f / 64.f);
  const float r = rsqrtf(ms + 1e-8f);
  outh[off] = (_Float16)((d * r * w[t] + b[t]) * (1.f - LAMBDA_INIT_F));
}

// ---------------------------------------- out = layernorm(a + res) (fp32 + opt f16)
__global__ __launch_bounds__(256)
void add_layernorm(const float* __restrict__ a, const float* __restrict__ res,
                   const float* __restrict__ w, const float* __restrict__ b,
                   float* __restrict__ out32, _Float16* __restrict__ out16) {
  const int s = blockIdx.x;
  const int t = threadIdx.x;
  __shared__ float sh[8];
  __shared__ float stat[2];
  const size_t base = (size_t)s * E_DIM;
  float v[4];
  float sum = 0.f;
  #pragma unroll
  for (int i = 0; i < 4; ++i) {
    const int e = t + i * 256;
    const float xv = a[base + e] + res[base + e];
    v[i] = xv; sum += xv;
  }
  #pragma unroll
  for (int o = 16; o > 0; o >>= 1) sum += __shfl_xor(sum, o, 32);
  if ((t & 31) == 0) sh[t >> 5] = sum;
  __syncthreads();
  if (t == 0) { float r = 0.f; for (int i = 0; i < 8; ++i) r += sh[i]; stat[0] = r * (1.f / E_DIM); }
  __syncthreads();
  const float mean = stat[0];
  float sq = 0.f;
  #pragma unroll
  for (int i = 0; i < 4; ++i) { const float dd = v[i] - mean; sq += dd * dd; }
  #pragma unroll
  for (int o = 16; o > 0; o >>= 1) sq += __shfl_xor(sq, o, 32);
  if ((t & 31) == 0) sh[t >> 5] = sq;
  __syncthreads();
  if (t == 0) { float r = 0.f; for (int i = 0; i < 8; ++i) r += sh[i]; stat[1] = r * (1.f / E_DIM); }
  __syncthreads();
  const float rstd = rsqrtf(stat[1] + 1e-5f);
  #pragma unroll
  for (int i = 0; i < 4; ++i) {
    const int e = t + i * 256;
    const float ov = (v[i] - mean) * rstd * w[e] + b[e];
    out32[base + e] = ov;
    if (out16) out16[base + e] = (_Float16)ov;
  }
}

// ------------------------------------------------------------------ launcher
extern "C" void kernel_launch(void* const* d_in, const int* in_sizes, int n_in,
                              void* d_out, int out_size, void* d_ws, size_t ws_size,
                              hipStream_t stream) {
  (void)in_sizes; (void)n_in; (void)out_size; (void)ws_size;
  const float* x      = (const float*)d_in[0];
  const float* emb    = (const float*)d_in[1];
  const float* qkv_w  = (const float*)d_in[2];
  const float* out_w  = (const float*)d_in[3];
  const float* sigmaw = (const float*)d_in[4];
  const float* lq1    = (const float*)d_in[5];
  const float* lk1    = (const float*)d_in[6];
  const float* lq2    = (const float*)d_in[7];
  const float* lk2    = (const float*)d_in[8];
  const float* ln_w   = (const float*)d_in[9];
  const float* ln_b   = (const float*)d_in[10];
  const float* ln1_w  = (const float*)d_in[11];
  const float* ln1_b  = (const float*)d_in[12];
  const float* ln2_w  = (const float*)d_in[13];
  const float* ln2_b  = (const float*)d_in[14];
  const float* w1     = (const float*)d_in[15];
  const float* b1     = (const float*)d_in[16];
  const float* w2     = (const float*)d_in[17];
  const float* b2     = (const float*)d_in[18];
  float* out = (float*)d_out;

  char* ws = (char*)d_ws;
  size_t off = 0;
  auto alloc = [&](size_t bytes) -> char* {
    char* p = ws + off;
    off += (bytes + 255) & ~(size_t)255;
    return p;
  };
  float*    sig   = (float*)alloc(D_DIM * sizeof(float));
  float*    lam   = (float*)alloc(sizeof(float));
  _Float16* xb    = (_Float16*)alloc((size_t)S_LEN * E_DIM * 2);
  _Float16* wqkvb = (_Float16*)alloc((size_t)3 * E_DIM * E_DIM * 2);
  _Float16* woutb = (_Float16*)alloc((size_t)E_DIM * E_DIM * 2);
  _Float16* w1b   = (_Float16*)alloc((size_t)4 * E_DIM * E_DIM * 2);
  _Float16* w2b   = (_Float16*)alloc((size_t)4 * E_DIM * E_DIM * 2);
  float*    qkvf  = (float*)alloc((size_t)S_LEN * 3 * E_DIM * 4);       // reused below
  _Float16* q1b   = (_Float16*)alloc((size_t)H_NUM * S_LEN * D_DIM * 2);
  _Float16* q2b   = (_Float16*)alloc((size_t)H_NUM * S_LEN * D_DIM * 2);
  _Float16* k1b   = (_Float16*)alloc((size_t)H_NUM * S_LEN * D_DIM * 2);
  _Float16* k2b   = (_Float16*)alloc((size_t)H_NUM * S_LEN * D_DIM * 2);
  _Float16* vtb   = (_Float16*)alloc((size_t)H_NUM * S_LEN * 2 * D_DIM * 2);  // [h][vdim][s]
  _Float16* abh   = (_Float16*)alloc((size_t)S_LEN * E_DIM * 2);
  float*    aof   = (float*)alloc((size_t)S_LEN * E_DIM * 4);           // reused below
  float*    hf    = (float*)alloc((size_t)S_LEN * E_DIM * 4);
  _Float16* hb16  = (_Float16*)alloc((size_t)S_LEN * E_DIM * 2);

  // aliases into dead regions (stream-ordered, deterministic)
  float*    attn1f = qkvf;                                  // qkv dead after rope
  float*    attn2f = qkvf + (size_t)S_LEN * E_DIM;
  _Float16* f1b    = (_Float16*)qkvf;                       // attn bufs dead after diffnorm
  float*    fff    = aof;                                   // aof dead after first add_layernorm

  // 1. fp32 -> f16 conversions (activations + weights)
  cvt_f32_f16<<<8192, 256, 0, stream>>>(x,     xb,    (long long)S_LEN * E_DIM);
  cvt_f32_f16<<<8192, 256, 0, stream>>>(qkv_w, wqkvb, (long long)3 * E_DIM * E_DIM);
  cvt_f32_f16<<<4096, 256, 0, stream>>>(out_w, woutb, (long long)E_DIM * E_DIM);
  cvt_f32_f16<<<8192, 256, 0, stream>>>(w1,    w1b,   (long long)4 * E_DIM * E_DIM);
  cvt_f32_f16<<<8192, 256, 0, stream>>>(w2,    w2b,   (long long)4 * E_DIM * E_DIM);
  small_setup<<<1, 64, 0, stream>>>(emb, sigmaw, lq1, lk1, lq2, lk2, sig, lam);

  // 2. qkv = x @ qkv_w^T  (2048x3072x1024)
  gemm_wmma<0, false><<<dim3(3 * E_DIM / BN, S_LEN / BM), 256, 0, stream>>>(
      xb, wqkvb, nullptr, qkvf, nullptr, S_LEN, 3 * E_DIM, E_DIM);
  // 3. RoPE + sigma + head split (V transposed)
  rope_split<<<S_LEN, 256, 0, stream>>>(qkvf, sig, q1b, q2b, k1b, k2b, vtb);
  // 4. two causal flash attentions (WMMA QK^T and PV)
  attn_flash<<<dim3(S_LEN / 64, H_NUM, 2), 128, 0, stream>>>(
      q1b, q2b, k1b, k2b, vtb, attn1f, attn2f, 0.17677669529663687f);
  // 5. per-head rmsnorm of attn1 - lam*attn2
  diff_rmsnorm<<<S_LEN * H_NUM, 64, 0, stream>>>(attn1f, attn2f, lam, ln_w, ln_b, abh);
  // 6. out projection (2048x1024x1024)
  gemm_wmma<0, false><<<dim3(E_DIM / BN, S_LEN / BM), 256, 0, stream>>>(
      abh, woutb, nullptr, aof, nullptr, S_LEN, E_DIM, E_DIM);
  // 7. h = layernorm(a + x)
  add_layernorm<<<S_LEN, 256, 0, stream>>>(aof, x, ln1_w, ln1_b, hf, hb16);
  // 8. ff1 = silu(h @ w1^T + b1) -> f16 (2048x4096x1024)
  gemm_wmma<1, true><<<dim3(4 * E_DIM / BN, S_LEN / BM), 256, 0, stream>>>(
      hb16, w1b, b1, nullptr, f1b, S_LEN, 4 * E_DIM, E_DIM);
  // 9. ff = ff1 @ w2^T + b2 (2048x1024x4096)
  gemm_wmma<0, true><<<dim3(E_DIM / BN, S_LEN / BM), 256, 0, stream>>>(
      f1b, w2b, b2, fff, nullptr, S_LEN, E_DIM, 4 * E_DIM);
  // 10. out = layernorm(ff + h)
  add_layernorm<<<S_LEN, 256, 0, stream>>>(fff, hf, ln2_w, ln2_b, out, nullptr);
}
